// multi_head_attention_51659866636405
// MI455X (gfx1250) — matrix operand order repaired and checked by reading, not compiled, not run on hardware
//
#include <hip/hip_runtime.h>
#include <math.h>

typedef __bf16 bf16_t;
typedef bf16_t  bf16x16 __attribute__((ext_vector_type(16)));
typedef float   floatx8 __attribute__((ext_vector_type(8)));

union FragB16 { bf16x16 v; unsigned int u[8]; bf16_t e[16]; };
union Pack4   { bf16_t e[4]; uint2 u; };

#define D_MODEL 1024
#define N_HEAD  16
#define HEAD_D  64
#define SEQ_T   2048
#define BATCH   2

static __device__ inline floatx8 zero8() {
    floatx8 z;
#pragma unroll
    for (int i = 0; i < 8; i++) z[i] = 0.0f;
    return z;
}

// CDNA5 async memory->LDS copy, 16B per lane, tracked by ASYNCcnt.
static __device__ inline void async_b128(unsigned lds_byte_off, const void* gaddr) {
    asm volatile("global_load_async_to_lds_b128 %0, %1, off"
                 :: "v"(lds_byte_off), "v"((unsigned long long)(uintptr_t)gaddr)
                 : "memory");
}
// Generic->LDS offset: low 32 bits of a generic LDS pointer are the DS address.
static __device__ inline unsigned lds_off(const void* p) {
    return (unsigned)(uintptr_t)p;
}

// ---------------- f32 -> bf16 pre-conversion ----------------
__global__ __launch_bounds__(256)
void cvt_f32_bf16_kernel(const float* __restrict__ src, bf16_t* __restrict__ dst, int n4)
{
    int i = blockIdx.x * blockDim.x + threadIdx.x;
    int stride = gridDim.x * blockDim.x;
    const float4* s4 = (const float4*)src;
    uint2* d2 = (uint2*)dst;
    for (; i < n4; i += stride) {
        float4 f = s4[i];
        Pack4 p;
        p.e[0] = (bf16_t)f.x; p.e[1] = (bf16_t)f.y;
        p.e[2] = (bf16_t)f.z; p.e[3] = (bf16_t)f.w;
        d2[i] = p.u;
    }
}

// C[M,N] = A[M,K] @ W[N,K]^T + bias, all-bf16 operands, f32 accumulate.
// mode 0: bf16 out scattered [B,H,T,Dh]; mode 1: f32 rowmajor [M,N];
// mode 2: bf16 out transposed [B,H,Dh,T] (for V).
__global__ __launch_bounds__(256)
void gemm_wmma_kernel(const bf16_t* __restrict__ A, const bf16_t* __restrict__ W,
                      const float* __restrict__ bias, void* __restrict__ outp,
                      int mode, int M, int N, int K)
{
    __shared__ __align__(16) bf16_t As[2][128 * 32];
    __shared__ __align__(16) bf16_t Bs[2][64 * 32];

    const int tid    = threadIdx.x;
    const int lane   = tid & 31;
    const int wave   = tid >> 5;        // 8 waves, 4x2 over (M,N)
    const int wm     = wave >> 1;
    const int wn     = wave & 1;
    const int m0     = blockIdx.y * 128;
    const int n0     = blockIdx.x * 64;
    const int laneN  = lane & 15;
    const int laneHi = (lane >> 4) & 1;

    floatx8 acc[2][2];
#pragma unroll
    for (int i = 0; i < 2; i++)
#pragma unroll
        for (int j = 0; j < 2; j++) acc[i][j] = zero8();

    // Issue one tile's async copies: A 128x32 (2 granules/thread) + W 64x32 (1/thread)
    auto issue_tile = [&](int k0, int bufi) {
#pragma unroll
        for (int i = 0; i < 2; i++) {
            int g = tid * 2 + i;              // 0..511
            int row = g >> 2, seg = g & 3;    // 4 x 16B segs per 32-col row
            async_b128(lds_off(&As[bufi][row * 32 + seg * 8]),
                       A + (size_t)(m0 + row) * K + k0 + seg * 8);
        }
        int n = tid >> 2, seg = tid & 3;
        async_b128(lds_off(&Bs[bufi][n * 32 + seg * 8]),
                   W + (size_t)(n0 + n) * K + k0 + seg * 8);
    };

    issue_tile(0, 0);
    int buf = 0;

    for (int k0 = 0; k0 < K; k0 += 32) {
        if (k0 + 32 < K) {
            issue_tile(k0 + 32, buf ^ 1);
            asm volatile("s_wait_asynccnt 0x3" ::: "memory");  // current tile landed
        } else {
            asm volatile("s_wait_asynccnt 0x0" ::: "memory");
        }
        __syncthreads();

        const unsigned int* AsU = (const unsigned int*)As[buf];
        const unsigned int* BsU = (const unsigned int*)Bs[buf];

        FragB16 afr[2], bfr[2];
#pragma unroll
        for (int mt = 0; mt < 2; mt++) {
            int row = wm * 32 + mt * 16 + laneN;       // A: lane holds row M=lane%16
#pragma unroll
            for (int r = 0; r < 8; r++)                // k = laneHi*16 + 2r
                afr[mt].u[r] = AsU[row * 16 + laneHi * 8 + r];
        }
#pragma unroll
        for (int nt = 0; nt < 2; nt++) {
            int nrow = wn * 32 + nt * 16 + laneN;      // B: lane holds col N=lane%16
#pragma unroll
            for (int r = 0; r < 8; r++)                // k = laneHi*16 + 2r
                bfr[nt].u[r] = BsU[nrow * 16 + laneHi * 8 + r];
        }
#pragma unroll
        for (int mt = 0; mt < 2; mt++)
#pragma unroll
            for (int nt = 0; nt < 2; nt++)
                acc[mt][nt] = __builtin_amdgcn_wmma_f32_16x16x32_bf16(
                    false, afr[mt].v, false, bfr[nt].v,
                    (short)0, acc[mt][nt], false, false);
        __syncthreads();
        buf ^= 1;
    }

    // epilogue: bias + store
#pragma unroll
    for (int mt = 0; mt < 2; mt++) {
#pragma unroll
        for (int nt = 0; nt < 2; nt++) {
            int col = n0 + wn * 32 + nt * 16 + laneN;
            float bv = bias[col];
#pragma unroll
            for (int r = 0; r < 8; r++) {
                int row = m0 + wm * 32 + mt * 16 + r + laneHi * 8;
                float val = acc[mt][nt][r] + bv;
                int b = row / SEQ_T, t = row % SEQ_T;
                int h = col >> 6, dh = col & 63;
                if (mode == 0) {
                    ((bf16_t*)outp)[(((size_t)(b * N_HEAD + h)) * SEQ_T + t) * HEAD_D + dh]
                        = (bf16_t)val;
                } else if (mode == 2) {
                    ((bf16_t*)outp)[(((size_t)(b * N_HEAD + h)) * HEAD_D + dh) * SEQ_T + t]
                        = (bf16_t)val;
                } else {
                    ((float*)outp)[(size_t)row * N + col] = val;
                }
            }
        }
    }
}

// Flash attention: block per (b,h,64 Q rows), 4 waves x 16 rows.
// K stored [B,H,T,Dh]; V stored transposed [B,H,Dh,T].
__global__ __launch_bounds__(128)
void attn_kernel(const bf16_t* __restrict__ Qp, const bf16_t* __restrict__ Kp,
                 const bf16_t* __restrict__ Vt, bf16_t* __restrict__ attn)
{
    __shared__ __align__(16) bf16_t Ks[2][32 * 64];   // [key][dh]
    __shared__ __align__(16) bf16_t Vs[2][64 * 32];   // [dh][key]
    __shared__ __align__(16) bf16_t Ps[4 * 16 * 32];

    const int tid    = threadIdx.x;
    const int lane   = tid & 31;
    const int wave   = tid >> 5;
    const int laneN  = lane & 15;
    const int laneHi = (lane >> 4) & 1;

    const int nQC = SEQ_T / 64;
    int bh  = blockIdx.x / nQC;
    int qc  = blockIdx.x % nQC;
    int b   = bh / N_HEAD, h = bh % N_HEAD;
    int qt0 = qc * 64 + wave * 16;

    // Async-stage one 32-key tile of K and V (4 granules/thread total)
    auto issue_tile = [&](int kt, int bufi) {
#pragma unroll
        for (int i = 0; i < 2; i++) {
            int g = tid * 2 + i;                 // 0..255
            int krow = g >> 3, kseg = g & 7;     // K: 32 rows x 8 segs
            async_b128(lds_off(&Ks[bufi][krow * 64 + kseg * 8]),
                       Kp + (((size_t)bh) * SEQ_T + kt + krow) * HEAD_D + kseg * 8);
            int vrow = g >> 2, vseg = g & 3;     // V^T: 64 dh rows x 4 segs
            async_b128(lds_off(&Vs[bufi][vrow * 32 + vseg * 8]),
                       Vt + (((size_t)bh) * HEAD_D + vrow) * SEQ_T + kt + vseg * 8);
        }
    };

    // Q fragments (16 rows x 64 dh = two A-frags), resident for whole loop
    FragB16 qf[2];
    {
        const unsigned int* Qg = (const unsigned int*)Qp;
        size_t base = (((size_t)bh) * SEQ_T + qt0 + laneN) * HEAD_D;
#pragma unroll
        for (int c = 0; c < 2; c++)
#pragma unroll
            for (int r = 0; r < 8; r++) {
                int k = c * 32 + laneHi * 16 + r * 2;
                qf[c].u[r] = Qg[(base + k) >> 1];
            }
    }

    floatx8 o[4];
#pragma unroll
    for (int c = 0; c < 4; c++) o[c] = zero8();
    float mi[8], li[8];
#pragma unroll
    for (int r = 0; r < 8; r++) { mi[r] = -__builtin_inff(); li[r] = 0.0f; }

    bf16_t* PsW = Ps + wave * 16 * 32;
    const unsigned int* PsU = (const unsigned int*)PsW;

    issue_tile(0, 0);
    int buf = 0;

    for (int kt = 0; kt < SEQ_T; kt += 32) {
        if (kt + 32 < SEQ_T) {
            issue_tile(kt + 32, buf ^ 1);
            asm volatile("s_wait_asynccnt 0x4" ::: "memory");
        } else {
            asm volatile("s_wait_asynccnt 0x0" ::: "memory");
        }
        __syncthreads();

        const unsigned int* KsU = (const unsigned int*)Ks[buf];
        const unsigned int* VsU = (const unsigned int*)Vs[buf];

        // S = Q @ K^T (16x32 = two 16x16 C tiles)
        floatx8 s0 = zero8(), s1 = zero8();
#pragma unroll
        for (int cg = 0; cg < 2; cg++) {
            FragB16 kf0, kf1;
            int key = cg * 16 + laneN;
#pragma unroll
            for (int r = 0; r < 8; r++) {
                kf0.u[r] = KsU[key * 32 + laneHi * 8 + r];        // dh 0..31
                kf1.u[r] = KsU[key * 32 + 16 + laneHi * 8 + r];   // dh 32..63
            }
            floatx8 s = zero8();
            s = __builtin_amdgcn_wmma_f32_16x16x32_bf16(false, qf[0].v, false, kf0.v,
                                                        (short)0, s, false, false);
            s = __builtin_amdgcn_wmma_f32_16x16x32_bf16(false, qf[1].v, false, kf1.v,
                                                        (short)0, s, false, false);
            if (cg == 0) s0 = s; else s1 = s;
        }

        const float scale = 0.125f;   // 1/sqrt(64)
#pragma unroll
        for (int r = 0; r < 8; r++) {
            float v0 = s0[r] * scale, v1 = s1[r] * scale;
            float mx = fmaxf(v0, v1);
#pragma unroll
            for (int msk = 1; msk < 16; msk <<= 1)
                mx = fmaxf(mx, __shfl_xor(mx, msk, 32));
            float mnew  = fmaxf(mi[r], mx);
            float alpha = __expf(mi[r] - mnew);
            float p0 = __expf(v0 - mnew), p1 = __expf(v1 - mnew);
            float sum = p0 + p1;
#pragma unroll
            for (int msk = 1; msk < 16; msk <<= 1)
                sum += __shfl_xor(sum, msk, 32);
            li[r] = li[r] * alpha + sum;
            mi[r] = mnew;
            o[0][r] *= alpha; o[1][r] *= alpha; o[2][r] *= alpha; o[3][r] *= alpha;
            int m = r + laneHi * 8;                 // C-layout -> LDS transpose staging
            PsW[m * 32 + laneN]      = (bf16_t)p0;
            PsW[m * 32 + 16 + laneN] = (bf16_t)p1;
        }
        __syncthreads();

        // O += P @ V : P re-read in A layout; V^T tile gives paired B reads
        FragB16 pf;
#pragma unroll
        for (int r = 0; r < 8; r++) {
            int k = laneHi * 16 + r * 2;
            pf.u[r] = PsU[(laneN * 32 + k) >> 1];
        }
#pragma unroll
        for (int c = 0; c < 4; c++) {
            FragB16 vf;
            int dh = c * 16 + laneN;
#pragma unroll
            for (int r = 0; r < 8; r++)             // keys (laneHi*16 + 2r, +1)
                vf.u[r] = VsU[dh * 16 + laneHi * 8 + r];
            o[c] = __builtin_amdgcn_wmma_f32_16x16x32_bf16(false, pf.v, false, vf.v,
                                                           (short)0, o[c], false, false);
        }
        __syncthreads();
        buf ^= 1;
    }

    // normalize + store merged-head bf16 [B,T,D]
#pragma unroll
    for (int r = 0; r < 8; r++) {
        int m = r + laneHi * 8;
        float inv = 1.0f / li[r];
        int t = qt0 + m;
        size_t base = ((size_t)b * SEQ_T + t) * D_MODEL + h * HEAD_D;
#pragma unroll
        for (int c = 0; c < 4; c++)
            attn[base + c * 16 + laneN] = (bf16_t)(o[c][r] * inv);
    }
}

extern "C" void kernel_launch(void* const* d_in, const int* in_sizes, int n_in,
                              void* d_out, int out_size, void* d_ws, size_t ws_size,
                              hipStream_t stream)
{
    const float* q   = (const float*)d_in[0];
    const float* k   = (const float*)d_in[1];
    const float* v   = (const float*)d_in[2];
    const float* w_q = (const float*)d_in[3];
    const float* b_q = (const float*)d_in[4];
    const float* w_k = (const float*)d_in[5];
    const float* b_k = (const float*)d_in[6];
    const float* w_v = (const float*)d_in[7];
    const float* b_v = (const float*)d_in[8];
    const float* w_o = (const float*)d_in[9];
    const float* b_o = (const float*)d_in[10];

    const int M = BATCH * SEQ_T;   // 4096
    const int N = D_MODEL;         // 1024
    const int K = D_MODEL;         // 1024
    const size_t ACT = (size_t)M * K * 2;       // 8 MiB bf16 activation
    const size_t WT  = (size_t)N * K * 2;       // 2 MiB bf16 weight

    char* ws = (char*)d_ws;
    bf16_t* qb   = (bf16_t*)(ws);
    bf16_t* kb   = (bf16_t*)(ws + ACT);
    bf16_t* vb   = (bf16_t*)(ws + 2 * ACT);
    bf16_t* wqb  = (bf16_t*)(ws + 3 * ACT);
    bf16_t* wkb  = (bf16_t*)(ws + 3 * ACT + WT);
    bf16_t* wvb  = (bf16_t*)(ws + 3 * ACT + 2 * WT);
    bf16_t* wob  = (bf16_t*)(ws + 3 * ACT + 3 * WT);
    bf16_t* Qp   = (bf16_t*)(ws + 3 * ACT + 4 * WT);
    bf16_t* Kp   = (bf16_t*)(ws + 4 * ACT + 4 * WT);
    bf16_t* Vtp  = (bf16_t*)(ws + 5 * ACT + 4 * WT);
    bf16_t* attn = (bf16_t*)(ws + 6 * ACT + 4 * WT);

    // f32 -> bf16 pre-conversion (vectorized, grid-stride)
    cvt_f32_bf16_kernel<<<1024, 256, 0, stream>>>(q,   qb,  M * K / 4);
    cvt_f32_bf16_kernel<<<1024, 256, 0, stream>>>(k,   kb,  M * K / 4);
    cvt_f32_bf16_kernel<<<1024, 256, 0, stream>>>(v,   vb,  M * K / 4);
    cvt_f32_bf16_kernel<<<512,  256, 0, stream>>>(w_q, wqb, N * K / 4);
    cvt_f32_bf16_kernel<<<512,  256, 0, stream>>>(w_k, wkb, N * K / 4);
    cvt_f32_bf16_kernel<<<512,  256, 0, stream>>>(w_v, wvb, N * K / 4);
    cvt_f32_bf16_kernel<<<512,  256, 0, stream>>>(w_o, wob, N * K / 4);

    dim3 gg(N / 64, M / 128);
    gemm_wmma_kernel<<<gg, 256, 0, stream>>>(qb, wqb, b_q, Qp,  0, M, N, K);
    gemm_wmma_kernel<<<gg, 256, 0, stream>>>(kb, wkb, b_k, Kp,  0, M, N, K);
    gemm_wmma_kernel<<<gg, 256, 0, stream>>>(vb, wvb, b_v, Vtp, 2, M, N, K);

    attn_kernel<<<BATCH * N_HEAD * (SEQ_T / 64), 128, 0, stream>>>(Qp, Kp, Vtp, attn);

    gemm_wmma_kernel<<<gg, 256, 0, stream>>>(attn, wob, b_o, d_out, 1, M, N, K);
}
